// DeepFM_3642132267189
// MI455X (gfx1250) — compile-verified, matrix operand order checked by reading, not run on hardware
//
#include <hip/hip_runtime.h>
#include <hip/hip_bf16.h>
#include <cstddef>
#include <cstdint>

// ---------------------------------------------------------------------------
// DeepFM forward for MI455X (gfx1250, wave32, WMMA).
// Pipeline: embed+FM -> weight bf16 transpose -> WMMA GEMM1 -> BN1 stats ->
//           BN1+ReLU->bf16 -> WMMA GEMM2 -> BN2 stats -> final (BN2+ReLU+W3+sigmoid)
//
// GEMM: block tile 128M x 128N, 8 waves as 4(M) x 2(N), wave tile 32M x 64N.
// B tile staged in LDS (double-buffered, async global->LDS when available),
// A fragments loaded directly from global (row-major K-major bf16).
// ---------------------------------------------------------------------------

typedef __attribute__((ext_vector_type(16))) __bf16 v16bf;
typedef __attribute__((ext_vector_type(8)))  float  v8f;
typedef __attribute__((ext_vector_type(4)))  unsigned int u32x4;
typedef int i32x4v __attribute__((vector_size(16)));   // matches builtin param type

#define BATCH   16384
#define F_CAT   26
#define F_CONT  13
#define VOCAB   100000
#define DIM     64
#define IN_DIM  2496      // (26+13)*64
#define H1      1024
#define H2      512
#define BN_EPS  1e-5f

// LDS row stride for the staged B tile: 32 K-elems (64 B) + 16 B pad
#define BROW_BYTES 80

#if __has_builtin(__builtin_amdgcn_global_load_async_to_lds_b128)
#define USE_ASYNC_LDS 1
#else
#define USE_ASYNC_LDS 0
#endif

union FragU {
    u32x4 u[2];
    v16bf v;
};

__device__ __forceinline__ void async_wait_all()
{
#if USE_ASYNC_LDS
#if __has_builtin(__builtin_amdgcn_s_wait_asynccnt)
    __builtin_amdgcn_s_wait_asynccnt(0);
#else
    asm volatile("s_wait_asynccnt 0" ::: "memory");
#endif
#endif
}

// ---------------------------------------------------------------------------
// Kernel 1: embeddings + FM first/second order. One block per batch row,
// 64 threads (one per embedding dim).
// ---------------------------------------------------------------------------
__global__ void embed_fm_kernel(const int* __restrict__ cat_feats,
                                const float* __restrict__ cont_feats,
                                const float* __restrict__ cat_t1,
                                const float* __restrict__ cat_t2,
                                const float* __restrict__ cont_t1,
                                const float* __restrict__ cont_t2,
                                __bf16* __restrict__ deep,
                                float* __restrict__ fm)
{
    const int row = blockIdx.x;
    const int d   = threadIdx.x;   // 0..63

    __shared__ int   idx[F_CAT];
    __shared__ float red[64];

    if (d < F_CAT) idx[d] = cat_feats[row * F_CAT + d];
    __syncthreads();

    __bf16* drow = deep + (size_t)row * IN_DIM;

    float s = 0.f, sq = 0.f;
    #pragma unroll
    for (int f = 0; f < F_CAT; ++f) {
        float e = cat_t2[((size_t)f * VOCAB + idx[f]) * DIM + d];
        s += e; sq += e * e;
        drow[f * DIM + d] = (__bf16)e;
    }
    #pragma unroll
    for (int f = 0; f < F_CONT; ++f) {
        float x = cont_feats[row * F_CONT + f];
        float e = cont_t2[f * DIM + d] * x;
        s += e; sq += e * e;
        drow[(F_CAT + f) * DIM + d] = (__bf16)e;
    }

    red[d] = 0.5f * (s * s - sq);
    __syncthreads();
    for (int st = 32; st > 0; st >>= 1) {
        if (d < st) red[d] += red[d + st];
        __syncthreads();
    }

    if (d == 0) {
        float ff = 0.f;
        #pragma unroll
        for (int f = 0; f < F_CAT; ++f)  ff += cat_t1[(size_t)f * VOCAB + idx[f]];
        #pragma unroll
        for (int f = 0; f < F_CONT; ++f) ff += cont_t1[f] * cont_feats[row * F_CONT + f];
        fm[row] = ff + red[0];
    }
}

// ---------------------------------------------------------------------------
// Kernel 2: transpose fp32 [K,N] -> bf16 [N,K] (both GEMM operands K-major,
// fragment loads become two contiguous b128 per lane).
// ---------------------------------------------------------------------------
__global__ void transpose_to_bf16(const float* __restrict__ W,
                                  __bf16* __restrict__ Wt,
                                  int K, int N)
{
    int i = blockIdx.x * blockDim.x + threadIdx.x;
    if (i >= K * N) return;
    int k = i / N;
    int n = i - k * N;
    Wt[(size_t)n * K + k] = (__bf16)W[i];
}

// ---------------------------------------------------------------------------
// WMMA bf16 GEMM:  C[M,N] = A[M,K](bf16 row-major) * Bt[N,K]^T + bias[N]
//
// Fragment addressing follows CDNA5 ISA 7.12.2 (16-bit A 16x32):
//   lanes 0-15  : row M=lane,    VGPR0-3 = K(kt+0..7),  VGPR4-7 = K(kt+16..23)
//   lanes 16-31 : row M=lane-16, VGPR0-3 = K(kt+8..15), VGPR4-7 = K(kt+24..31)
// B mirrors this with N on lanes; Bt is [N,K] so addressing is identical.
//
// B tile (128 N x 32 K) is staged in LDS per k-step, double-buffered,
// via GLOBAL_LOAD_ASYNC_TO_LDS_B128 (ASYNCcnt) when the builtin exists.
// ---------------------------------------------------------------------------
template<int K, int N>
__global__ __launch_bounds__(256) void gemm_bf16_wmma(
        const __bf16* __restrict__ A,
        const __bf16* __restrict__ Bt,
        const float* __restrict__ bias,
        float* __restrict__ C)
{
    const int tid  = threadIdx.x;
    const int wave = tid >> 5;
    const int lane = tid & 31;
    const int half = lane >> 4;      // 0: lanes 0-15, 1: lanes 16-31
    const int l16  = lane & 15;
    const int wm   = wave >> 1;      // 0..3 (M direction)
    const int wn   = wave & 1;       // 0..1 (N direction)

    const int m0  = blockIdx.y * 128 + wm * 32;   // wave's 32 rows
    const int n0b = blockIdx.x * 128;             // block's 128 cols
    const int n0w = wn * 64;                      // wave col offset in block

    // Double-buffered staged B tile: 128 rows x 80 B
    __shared__ alignas(16) char smem[2][128 * BROW_BYTES];

    const __bf16* arow0 = A + (size_t)(m0 + l16) * K;
    const __bf16* arow1 = A + (size_t)(m0 + 16 + l16) * K;

    // staging: 128 rows x 4 chunks(16B) = 512 chunks; 256 threads x 2
    const int ch0_row  = (tid * 2) >> 2;
    const int ch0_part = (tid * 2) & 3;
    const int ch1_row  = (tid * 2 + 1) >> 2;
    const int ch1_part = (tid * 2 + 1) & 3;
    const __bf16* bsrc0 = Bt + (size_t)(n0b + ch0_row) * K + ch0_part * 8;
    const __bf16* bsrc1 = Bt + (size_t)(n0b + ch1_row) * K + ch1_part * 8;
    const int bdst0 = ch0_row * BROW_BYTES + ch0_part * 16;
    const int bdst1 = ch1_row * BROW_BYTES + ch1_part * 16;

    auto stageB = [&](int buf, int kt) {
#if USE_ASYNC_LDS
        __builtin_amdgcn_global_load_async_to_lds_b128(
            (__attribute__((address_space(1))) i32x4v*)(bsrc0 + kt),
            (__attribute__((address_space(3))) i32x4v*)(&smem[buf][bdst0]), 0, 0);
        __builtin_amdgcn_global_load_async_to_lds_b128(
            (__attribute__((address_space(1))) i32x4v*)(bsrc1 + kt),
            (__attribute__((address_space(3))) i32x4v*)(&smem[buf][bdst1]), 0, 0);
#else
        *(u32x4*)(&smem[buf][bdst0]) = *(const u32x4*)(bsrc0 + kt);
        *(u32x4*)(&smem[buf][bdst1]) = *(const u32x4*)(bsrc1 + kt);
#endif
    };

    v8f acc[2][4];
    #pragma unroll
    for (int i = 0; i < 2; ++i)
        #pragma unroll
        for (int j = 0; j < 4; ++j)
            acc[i][j] = (v8f){};

    // prologue: stage first B slice
    stageB(0, 0);
    async_wait_all();
    __syncthreads();

    for (int kt = 0; kt < K; kt += 32) {
        const int cur = (kt >> 5) & 1;
        if (kt + 32 < K) stageB(cur ^ 1, kt + 32);

        // A fragments (direct from global)
        FragU a0, a1;
        a0.u[0] = *(const u32x4*)(arow0 + kt + half * 8);
        a0.u[1] = *(const u32x4*)(arow0 + kt + 16 + half * 8);
        a1.u[0] = *(const u32x4*)(arow1 + kt + half * 8);
        a1.u[1] = *(const u32x4*)(arow1 + kt + 16 + half * 8);

        // B fragments from LDS, 8 WMMAs
        #pragma unroll
        for (int j = 0; j < 4; ++j) {
            const char* base = &smem[cur][(n0w + j * 16 + l16) * BROW_BYTES + half * 16];
            FragU b;
            b.u[0] = *(const u32x4*)(base);
            b.u[1] = *(const u32x4*)(base + 32);
            acc[0][j] = __builtin_amdgcn_wmma_f32_16x16x32_bf16(
                false, a0.v, false, b.v, (short)0, acc[0][j], false, false);
            acc[1][j] = __builtin_amdgcn_wmma_f32_16x16x32_bf16(
                false, a1.v, false, b.v, (short)0, acc[1][j], false, false);
        }

        async_wait_all();
        __syncthreads();
    }

    // C/D layout (32-bit 16x16): lanes 0-15 N=lane M=r, lanes 16-31 N=lane-16 M=r+8
    #pragma unroll
    for (int i = 0; i < 2; ++i) {
        const int rbase = m0 + i * 16 + half * 8;
        #pragma unroll
        for (int j = 0; j < 4; ++j) {
            const int col = n0b + n0w + j * 16 + l16;
            const float bj = bias[col];
            #pragma unroll
            for (int r = 0; r < 8; ++r) {
                C[(size_t)(rbase + r) * N + col] = acc[i][j][r] + bj;
            }
        }
    }
}

// ---------------------------------------------------------------------------
// BN stats: one block per column; batch mean / biased var ->
// precomputed affine scale/shift.
// ---------------------------------------------------------------------------
__global__ void bn_stats_kernel(const float* __restrict__ C,
                                const float* __restrict__ g,
                                const float* __restrict__ be,
                                float* __restrict__ scale,
                                float* __restrict__ shift,
                                int Mrows, int Ncols)
{
    const int j = blockIdx.x;
    const int t = threadIdx.x;   // 256
    float s = 0.f, sq = 0.f;
    for (int r = t; r < Mrows; r += 256) {
        float x = C[(size_t)r * Ncols + j];
        s += x; sq += x * x;
    }
    __shared__ float rs[256], rq[256];
    rs[t] = s; rq[t] = sq;
    __syncthreads();
    for (int st = 128; st > 0; st >>= 1) {
        if (t < st) { rs[t] += rs[t + st]; rq[t] += rq[t + st]; }
        __syncthreads();
    }
    if (t == 0) {
        float inv = 1.0f / (float)Mrows;
        float mu  = rs[0] * inv;
        float var = rq[0] * inv - mu * mu;
        float rstd = rsqrtf(var + BN_EPS);
        float sc = g[j] * rstd;
        scale[j] = sc;
        shift[j] = be[j] - sc * mu;
    }
}

// ---------------------------------------------------------------------------
// BN apply + ReLU + convert to bf16
// ---------------------------------------------------------------------------
__global__ void bn_relu_bf16_kernel(const float* __restrict__ C,
                                    const float* __restrict__ scale,
                                    const float* __restrict__ shift,
                                    __bf16* __restrict__ H,
                                    int total, int Ncols)
{
    int i = blockIdx.x * blockDim.x + threadIdx.x;
    if (i >= total) return;
    int col = i % Ncols;
    float y = fmaf(C[i], scale[col], shift[col]);
    H[i] = (__bf16)fmaxf(y, 0.f);
}

// ---------------------------------------------------------------------------
// Final: BN2 + ReLU + dot(W3) + bias + fm -> sigmoid -> [1-p, p]
// One block per row, 64 threads.
// ---------------------------------------------------------------------------
__global__ void final_kernel(const float* __restrict__ C2,
                             const float* __restrict__ scale2,
                             const float* __restrict__ shift2,
                             const float* __restrict__ W3,
                             const float* __restrict__ b3,
                             const float* __restrict__ bias0,
                             const float* __restrict__ fm,
                             float* __restrict__ out)
{
    const int row = blockIdx.x;
    const int t   = threadIdx.x;  // 64
    float z = 0.f;
    #pragma unroll
    for (int j = t; j < H2; j += 64) {
        float y = fmaf(C2[(size_t)row * H2 + j], scale2[j], shift2[j]);
        y = fmaxf(y, 0.f);
        z += y * W3[j];
    }
    __shared__ float red[64];
    red[t] = z;
    __syncthreads();
    for (int st = 32; st > 0; st >>= 1) {
        if (t < st) red[t] += red[t + st];
        __syncthreads();
    }
    if (t == 0) {
        float logit = bias0[0] + fm[row] + red[0] + b3[0];
        float p = 1.f / (1.f + __expf(-logit));
        out[row * 2 + 0] = 1.f - p;
        out[row * 2 + 1] = p;
    }
}

// ---------------------------------------------------------------------------
// Launcher
// ---------------------------------------------------------------------------
extern "C" void kernel_launch(void* const* d_in, const int* in_sizes, int n_in,
                              void* d_out, int out_size, void* d_ws, size_t ws_size,
                              hipStream_t stream)
{
    (void)in_sizes; (void)n_in; (void)out_size; (void)ws_size;

    const int*   cat_feats  = (const int*)  d_in[0];
    const float* cont_feats = (const float*)d_in[1];
    const float* bias0      = (const float*)d_in[2];
    const float* cat_t1     = (const float*)d_in[3];
    const float* cat_t2     = (const float*)d_in[4];
    const float* cont_t1    = (const float*)d_in[5];
    const float* cont_t2    = (const float*)d_in[6];
    const float* W1         = (const float*)d_in[7];
    const float* b1         = (const float*)d_in[8];
    const float* g1         = (const float*)d_in[9];
    const float* be1        = (const float*)d_in[10];
    const float* W2         = (const float*)d_in[11];
    const float* b2         = (const float*)d_in[12];
    const float* g2         = (const float*)d_in[13];
    const float* be2        = (const float*)d_in[14];
    const float* W3         = (const float*)d_in[15];
    const float* b3         = (const float*)d_in[16];

    float* out = (float*)d_out;

    // --- workspace layout (all sizes multiples of 256 B) ---
    char* ws = (char*)d_ws;
    size_t cur = 0;
    auto alloc = [&](size_t bytes) { char* p = ws + cur; cur += bytes; return p; };

    __bf16* deep  = (__bf16*)alloc((size_t)BATCH * IN_DIM * sizeof(__bf16)); // 81.8 MB
    float*  fm    = (float*) alloc((size_t)BATCH * sizeof(float));
    __bf16* W1t   = (__bf16*)alloc((size_t)H1 * IN_DIM * sizeof(__bf16));    // 5.1 MB
    __bf16* W2t   = (__bf16*)alloc((size_t)H2 * H1 * sizeof(__bf16));        // 1.0 MB
    float*  C1    = (float*) alloc((size_t)BATCH * H1 * sizeof(float));      // 67 MB
    __bf16* h1b   = (__bf16*)alloc((size_t)BATCH * H1 * sizeof(__bf16));     // 33.5 MB
    float*  C2    = (float*) alloc((size_t)BATCH * H2 * sizeof(float));      // 33.5 MB
    float*  sc1   = (float*) alloc(H1 * sizeof(float));
    float*  sh1   = (float*) alloc(H1 * sizeof(float));
    float*  sc2   = (float*) alloc(H2 * sizeof(float));
    float*  sh2   = (float*) alloc(H2 * sizeof(float));

    // 1) embeddings + FM terms
    embed_fm_kernel<<<BATCH, 64, 0, stream>>>(cat_feats, cont_feats, cat_t1, cat_t2,
                                              cont_t1, cont_t2, deep, fm);

    // 2) weight prep (fp32 [K,N] -> bf16 [N,K])
    {
        int n1 = IN_DIM * H1;
        transpose_to_bf16<<<(n1 + 255) / 256, 256, 0, stream>>>(W1, W1t, IN_DIM, H1);
        int n2 = H1 * H2;
        transpose_to_bf16<<<(n2 + 255) / 256, 256, 0, stream>>>(W2, W2t, H1, H2);
    }

    // 3) GEMM1: C1 = deep @ W1 + b1   (block 128x128)
    gemm_bf16_wmma<IN_DIM, H1><<<dim3(H1 / 128, BATCH / 128), 256, 0, stream>>>(
        deep, W1t, b1, C1);

    // 4) BN1 stats + apply (ReLU, bf16)
    bn_stats_kernel<<<H1, 256, 0, stream>>>(C1, g1, be1, sc1, sh1, BATCH, H1);
    {
        int total = BATCH * H1;
        bn_relu_bf16_kernel<<<(total + 255) / 256, 256, 0, stream>>>(C1, sc1, sh1,
                                                                     h1b, total, H1);
    }

    // 5) GEMM2: C2 = h1 @ W2 + b2
    gemm_bf16_wmma<H1, H2><<<dim3(H2 / 128, BATCH / 128), 256, 0, stream>>>(
        h1b, W2t, b2, C2);

    // 6) BN2 stats
    bn_stats_kernel<<<H2, 256, 0, stream>>>(C2, g2, be2, sc2, sh2, BATCH, H2);

    // 7) final: BN2+ReLU -> W3 dot -> sigmoid -> [1-p, p]
    final_kernel<<<BATCH, 64, 0, stream>>>(C2, sc2, sh2, W3, b3, bias0, fm, out);
}
